// LogEig_12893491822887
// MI455X (gfx1250) — matrix operand order repaired and checked by reading, not compiled, not run on hardware
//
#include <hip/hip_runtime.h>
#include <math.h>

// ---------------------------------------------------------------------------
// Batched SPD matrix logarithm, 16384 x (64x64) fp32.
//   log(X) = 2^k * log((X/c)^(1/2^k)) + log(c) I,   c = ||X||_F
// Square roots via coupled Newton-Schulz (GEMM-only), log via Horner series.
// All iterates are polynomials of the symmetric input -> symmetric, so the
// B-operand of every GEMM is read row-wise from the same row-major storage.
// Matrices live in LDS as split-bf16 planes (hi+lo ~ fp32 accuracy); GEMMs
// use the 3xBF16 trick on v_wmma_f32_16x16x32_bf16 with fp32 accumulate.
// Fragment feeds are pure ds_load_b128, conversion happens once per output.
// ---------------------------------------------------------------------------

typedef __bf16 bf16_t;
typedef __attribute__((ext_vector_type(16))) __bf16 v16bf;
typedef __attribute__((ext_vector_type(8)))  __bf16 v8bf;
typedef __attribute__((ext_vector_type(8)))  float  v8f;

#define MATN   64
#define STR    72          // bf16 row stride: 144 B, 16B-aligned, conflict-free b128
#define NSQRT  5           // 2^5 = 32
#define MTERMS 8           // log-series terms

__device__ __forceinline__ v16bf concat8(v8bf a, v8bf b) {
  return __builtin_shufflevector(a, b, 0, 1, 2, 3, 4, 5, 6, 7,
                                       8, 9, 10, 11, 12, 13, 14, 15);
}

// A fragment (16x32 bf16): row = tile_row*16 + (lane&15);
// K(e) = kbase + e + 8h (e<8), kbase + 16 + (e-8) + 8h (e>=8)
// -> two contiguous 16B chunks.
__device__ __forceinline__ v16bf loadA(const bf16_t* P, int row, int kbase, int h) {
  const bf16_t* base = P + row * STR + kbase + 8 * h;
  v8bf c0 = *(const v8bf*)(base);
  v8bf c1 = *(const v8bf*)(base + 16);
  return concat8(c0, c1);
}

// B fragment (32x16 bf16): K(e) = kbase + e + 8h (e<8), kbase + 16 + (e-8) + 8h (e>=8) at column n.
// Symmetric matrix: column n == row n -> two contiguous 16B chunks.
__device__ __forceinline__ v16bf loadB(const bf16_t* P, int col, int kbase, int h) {
  const bf16_t* base = P + col * STR + kbase + 8 * h;
  v8bf c0 = *(const v8bf*)(base);
  v8bf c1 = *(const v8bf*)(base + 16);
  return concat8(c0, c1);
}

__device__ __forceinline__ void splitStore(bf16_t* hi, bf16_t* lo, int idx, float v) {
  bf16_t h = (bf16_t)v;
  hi[idx] = h;
  lo[idx] = (bf16_t)(v - (float)h);
}

__device__ __forceinline__ float joinLoad(const bf16_t* hi, const bf16_t* lo, int idx) {
  return (float)hi[idx] + (float)lo[idx];
}

// Core tile compute: acc(8xf32) for output tile (ti,tj) of alpha*(A@B)
__device__ __forceinline__ v8f tileMM(const bf16_t* Ahi, const bf16_t* Alo,
                                      const bf16_t* Bhi, const bf16_t* Blo,
                                      int ti, int tj, int m15, int h) {
  v8f acc = {};
#pragma unroll
  for (int ks = 0; ks < 2; ++ks) {
    const int kbase = ks * 32;
    v16bf ah = loadA(Ahi, ti * 16 + m15, kbase, h);
    v16bf al = loadA(Alo, ti * 16 + m15, kbase, h);
    v16bf bh = loadB(Bhi, tj * 16 + m15, kbase, h);
    v16bf bl = loadB(Blo, tj * 16 + m15, kbase, h);
    acc = __builtin_amdgcn_wmma_f32_16x16x32_bf16(false, ah, false, bh,
                                                  (short)0, acc, false, false);
    acc = __builtin_amdgcn_wmma_f32_16x16x32_bf16(false, ah, false, bl,
                                                  (short)0, acc, false, false);
    acc = __builtin_amdgcn_wmma_f32_16x16x32_bf16(false, al, false, bh,
                                                  (short)0, acc, false, false);
  }
  return acc;
}

// D = alpha*(A@B) + diag*I, split-bf16 planes in LDS. 8 waves x 2 tiles.
__device__ __forceinline__ void gemm64(bf16_t* Dhi, bf16_t* Dlo,
                                       const bf16_t* Ahi, const bf16_t* Alo,
                                       const bf16_t* Bhi, const bf16_t* Blo,
                                       float alpha, float diag,
                                       int lane, int wave) {
  const int m15 = lane & 15;
  const int h   = lane >> 4;
#pragma unroll
  for (int t = 0; t < 2; ++t) {
    const int tile = wave * 2 + t;
    const int ti = tile >> 2, tj = tile & 3;
    v8f acc = tileMM(Ahi, Alo, Bhi, Blo, ti, tj, m15, h);
#pragma unroll
    for (int v = 0; v < 8; ++v) {
      const int r = ti * 16 + v + 8 * h;
      const int c = tj * 16 + m15;
      const float val = alpha * acc[v] + ((r == c) ? diag : 0.0f);
      splitStore(Dhi, Dlo, r * STR + c, val);
    }
  }
}

// Final fused GEMM: out(global,fp32) = scale*(A@B) + logc*I
__device__ __forceinline__ void gemm64_out(float* __restrict__ outb,
                                           const bf16_t* Ahi, const bf16_t* Alo,
                                           const bf16_t* Bhi, const bf16_t* Blo,
                                           float scale, float logc,
                                           int lane, int wave) {
  const int m15 = lane & 15;
  const int h   = lane >> 4;
#pragma unroll
  for (int t = 0; t < 2; ++t) {
    const int tile = wave * 2 + t;
    const int ti = tile >> 2, tj = tile & 3;
    v8f acc = tileMM(Ahi, Alo, Bhi, Blo, ti, tj, m15, h);
#pragma unroll
    for (int v = 0; v < 8; ++v) {
      const int r = ti * 16 + v + 8 * h;
      const int c = tj * 16 + m15;
      outb[r * MATN + c] = scale * acc[v] + ((r == c) ? logc : 0.0f);
    }
  }
}

__global__ __launch_bounds__(256)
void logm_spd_kernel(const float* __restrict__ X, float* __restrict__ out, int batch) {
  // 4 matrices x (hi,lo) planes: 4 * 2 * 64*72*2B = 72 KB of 320 KB WGP LDS
  __shared__ __align__(16) bf16_t bufs[4][2][MATN * STR];
  __shared__ float red[256];
  __shared__ float sc[2];

  const int tid  = threadIdx.x;
  const int lane = tid & 31;
  const int wave = tid >> 5;
  const int b    = blockIdx.x;
  if (b >= batch) return;  // uniform

  const float* Xb = X + (size_t)b * (MATN * MATN);

  bf16_t *Ah = bufs[0][0], *Al = bufs[0][1];   // current matrix / Y
  bf16_t *Zh = bufs[1][0], *Zl = bufs[1][1];
  bf16_t *Th = bufs[2][0], *Tl = bufs[2][1];
  bf16_t *Wh = bufs[3][0], *Wl = bufs[3][1];

  // ---- pass 1: Frobenius norm of X ----
  float ss = 0.0f;
  for (int e = tid; e < MATN * MATN; e += 256) {
    const float x = Xb[e];
    ss += x * x;
  }
  red[tid] = ss;
  __syncthreads();
  for (int off = 128; off; off >>= 1) {
    if (tid < off) red[tid] += red[tid + off];
    __syncthreads();
  }
  if (tid == 0) {
    const float c = fmaxf(sqrtf(red[0]), 1e-20f);
    sc[0] = 1.0f / c;
    sc[1] = logf(c);
  }
  __syncthreads();
  const float invc = sc[0];
  const float logc = sc[1];

  // ---- pass 2 (L2-hot): A0 = X/c into split planes ----
  for (int e = tid; e < MATN * MATN; e += 256) {
    const int r = e >> 6, c = e & 63;
    splitStore(Ah, Al, r * STR + c, Xb[e] * invc);
  }
  __syncthreads();

  // ---- k repeated square roots via coupled Newton-Schulz ----
  const int ns_sched[NSQRT] = {16, 12, 10, 9, 8};
  for (int s = 0; s < NSQRT; ++s) {
    for (int e = tid; e < MATN * MATN; e += 256) {   // Z = I
      const int r = e >> 6, c = e & 63;
      splitStore(Zh, Zl, r * STR + c, (r == c) ? 1.0f : 0.0f);
    }
    __syncthreads();
    const int nns = ns_sched[s];
    for (int it = 0; it < nns; ++it) {
      gemm64(Th, Tl, Zh, Zl, Ah, Al, -0.5f, 1.5f, lane, wave);  // T = 1.5I - 0.5 Z Y
      __syncthreads();
      gemm64(Wh, Wl, Ah, Al, Th, Tl, 1.0f, 0.0f, lane, wave);   // Ynew = Y T
      __syncthreads();
      gemm64(Ah, Al, Th, Tl, Zh, Zl, 1.0f, 0.0f, lane, wave);   // Znew = T Z
      __syncthreads();
      bf16_t* oh = Zh; bf16_t* ol = Zl;
      Zh = Ah; Zl = Al;     // Znew
      Ah = Wh; Al = Wl;     // Ynew
      Wh = oh; Wl = ol;     // recycled
    }
  }

  // ---- E = A - I  (diagonal adjust only) ----
  if (tid < MATN) {
    const int idx = tid * STR + tid;
    splitStore(Ah, Al, idx, joinLoad(Ah, Al, idx) - 1.0f);
  }
  __syncthreads();

  // ---- log(I+E) Horner: P_m = c_m I; P_j = E@P + c_j I; out = 2^k (E@P_1) + logc I
  {
    const float cm = (((MTERMS & 1) ? 1.0f : -1.0f) / (float)MTERMS);
    for (int e = tid; e < MATN * MATN; e += 256) {
      const int r = e >> 6, c = e & 63;
      splitStore(Zh, Zl, r * STR + c, (r == c) ? cm : 0.0f);
    }
  }
  __syncthreads();
  bf16_t *Ph = Zh, *Pl = Zl, *Qh = Wh, *Ql = Wl;
  for (int j = MTERMS - 1; j >= 1; --j) {
    const float cj = (((j & 1) ? 1.0f : -1.0f) / (float)j);
    gemm64(Qh, Ql, Ah, Al, Ph, Pl, 1.0f, cj, lane, wave);
    __syncthreads();
    bf16_t* th = Ph; bf16_t* tl = Pl;
    Ph = Qh; Pl = Ql; Qh = th; Ql = tl;
  }
  // fused final: out = 2^k * (E @ P1) + logc * I
  gemm64_out(out + (size_t)b * (MATN * MATN),
             Ah, Al, Ph, Pl, (float)(1 << NSQRT), logc, lane, wave);
}

extern "C" void kernel_launch(void* const* d_in, const int* in_sizes, int n_in,
                              void* d_out, int out_size, void* d_ws, size_t ws_size,
                              hipStream_t stream) {
  (void)n_in; (void)out_size; (void)d_ws; (void)ws_size;
  const float* X = (const float*)d_in[0];
  float* out = (float*)d_out;
  const int batch = in_sizes[0] / (MATN * MATN);   // 16384
  dim3 grid(batch), block(256);
  hipLaunchKernelGGL(logm_spd_kernel, grid, block, 0, stream, X, out, batch);
}
